// att_lstm_2405181686385
// MI455X (gfx1250) — compile-verified, matrix operand order checked by reading, not run on hardware
//
#include <hip/hip_runtime.h>

// ---------------------------------------------------------------------------
// CDNA5 (gfx1250) wave32 WMMA implementation of the attention-LSTM ranker.
// bf16 WMMA for all large GEMMs + LSTM recurrence; fp32 accumulation.
// Global->LDS tile staging via the Tensor Data Mover (tensor_load_to_lds),
// double-buffered, synchronized with s_wait_tensorcnt.
// ---------------------------------------------------------------------------

typedef __attribute__((ext_vector_type(16))) __bf16 v16bf;
typedef __attribute__((ext_vector_type(8)))  float  v8f;
typedef unsigned int __attribute__((ext_vector_type(4))) tdm_u32x4;
typedef int          __attribute__((ext_vector_type(8))) tdm_i32x8;
typedef int          __attribute__((ext_vector_type(4))) tdm_i32x4;

union BF16x16 { v16bf v; unsigned short u[16]; };
union F32x8   { v8f v; float f[8]; };

__device__ __forceinline__ unsigned short f2bfu(float f) {
  unsigned u = __builtin_bit_cast(unsigned, f);
  u += 0x7FFFu + ((u >> 16) & 1u);            // round-to-nearest-even
  return (unsigned short)(u >> 16);
}
__device__ __forceinline__ float bfu2f(unsigned short s) {
  unsigned u = ((unsigned)s) << 16;
  return __builtin_bit_cast(float, u);
}
__device__ __forceinline__ float sigmoidf(float x) {
  return 1.0f / (1.0f + expf(-x));
}
// LDS aperture: flat address low 32 bits are the LDS byte address.
__device__ __forceinline__ unsigned lds_off(const void* p) {
  return (unsigned)(unsigned long long)p;
}

// ---------------------------------------------------------------------------
// TDM: 2D bf16 tile (tile_d0 x tile_d1 elements, row stride stride0 elements)
// from global to LDS, with LDS padding pad_amount DWORDs every pad_interval
// DWORDs (codes per D# group1). D# packing per CDNA5 ISA 8.3-8.6.
// ---------------------------------------------------------------------------
__device__ __forceinline__ void tdm_load_2d_bf16(
    unsigned ldsaddr, const void* gptr, unsigned tile_d0, unsigned tile_d1,
    unsigned long long tensor_d0, unsigned long long tensor_d1,
    unsigned long long stride0, int pad_interval_code, int pad_amount_code) {
  unsigned long long ga = (unsigned long long)gptr;
  tdm_u32x4 g0;
  g0[0] = 1u;                                        // count=1 (valid), user mode
  g0[1] = ldsaddr;                                   // lds_addr [63:32]
  g0[2] = (unsigned)(ga & 0xFFFFFFFFu);              // global_addr [95:64]
  g0[3] = (unsigned)((ga >> 32) & 0x01FFFFFFu) | (2u << 30);  // ga hi | type=2
  tdm_i32x8 g1;
  unsigned w0 = (1u << 16);                          // data_size=1 (2 bytes)
  if (pad_amount_code >= 0)
    w0 |= (1u << 20) | ((unsigned)pad_interval_code << 22) |
          ((unsigned)pad_amount_code << 25);         // pad_enable|interval|amount
  g1[0] = (int)w0;                                   // wg_mask=0
  g1[1] = (int)((unsigned)(tensor_d0 & 0xFFFFu) << 16);            // dim0 lo16
  g1[2] = (int)(((tensor_d0 >> 16) & 0xFFFFu) |
                ((tensor_d1 & 0xFFFFu) << 16));                    // dim0 hi|dim1 lo
  g1[3] = (int)(((tensor_d1 >> 16) & 0xFFFFu) | (tile_d0 << 16));  // dim1 hi|tile0
  g1[4] = (int)(tile_d1 & 0xFFFFu);                                // tile1, tile2=0
  g1[5] = (int)(stride0 & 0xFFFFFFFFu);                            // stride0 lo32
  g1[6] = (int)((stride0 >> 32) & 0xFFFFu);                        // stride0 hi16
  g1[7] = 0;                                                       // stride1=0
  tdm_i32x4 g2 = {0, 0, 0, 0};
  tdm_i32x4 g3 = {0, 0, 0, 0};
#if __clang_major__ >= 23
  tdm_i32x8 g4 = {0, 0, 0, 0, 0, 0, 0, 0};
  __builtin_amdgcn_tensor_load_to_lds(g0, g1, g2, g3, g4, 0);
#else
  __builtin_amdgcn_tensor_load_to_lds(g0, g1, g2, g3, 0);
#endif
}

// ---------------------------------------------------------------------------
// fp32 (K x N) -> bf16 transposed (N x K) weight conversion
// ---------------------------------------------------------------------------
__global__ void cvt_bf16_t_kernel(const float* __restrict__ src,
                                  unsigned short* __restrict__ dst,
                                  int K, int N) {
  int i = blockIdx.x * 256 + threadIdx.x;
  if (i < N * K) {
    int n = i / K, k = i - n * K;
    dst[i] = f2bfu(src[(long)k * N + n]);
  }
}

// ---------------------------------------------------------------------------
// Tiled WMMA GEMM: out[M,N] = act(A[M,K] @ W[K,N] + bias), W given as Wt[N,K].
//   AMODE 0: A gathered fp32 from 5 concatenated (B,S,256) tensors -> bf16
//   AMODE 1: A bf16 row-major, staged by TDM
//   OMODE 0: relu -> bf16      OMODE 1: none -> fp32
// Block: 256 threads (8 waves). Tile 128x64, K-chunk 32, double-buffered TDM.
// Wave (wm in [0,4), wn in [0,2)): 32 rows x 32 cols = 2x2 C-fragments.
// ---------------------------------------------------------------------------
template <int AMODE, int OMODE>
__global__ __launch_bounds__(256) void gemm_bf16_kernel(
    const float* __restrict__ a0, const float* __restrict__ a1,
    const float* __restrict__ a2, const float* __restrict__ a3,
    const float* __restrict__ a4,
    const unsigned short* __restrict__ Abf,
    const unsigned short* __restrict__ Wt,      // (N x K) bf16
    const float* __restrict__ bias,
    unsigned short* __restrict__ outBf, float* __restrict__ outF,
    int M, int N, int K) {
  __shared__ unsigned short Asl[2][128][40];    // [m][k], pitch 40 (TDM pad)
  __shared__ unsigned short Bsl[2][64][40];     // [n][k], pitch 40 (TDM pad)

  const int tid = threadIdx.x;
  const int nbN = N >> 6;
  const int mb = blockIdx.x / nbN;
  const int nb = blockIdx.x % nbN;
  const int wave = tid >> 5, lane = tid & 31;
  const int wm = wave >> 1, wn = wave & 1;
  const int lrow = lane & 15;
  const int lhalf = (lane >> 4) * 8;

  F32x8 acc[2][2];
#pragma unroll
  for (int mt = 0; mt < 2; ++mt)
#pragma unroll
    for (int nt = 0; nt < 2; ++nt)
#pragma unroll
      for (int i = 0; i < 8; ++i) acc[mt][nt].f[i] = 0.0f;

  const float* srcs[5] = {a0, a1, a2, a3, a4};
  const int arow = tid >> 1;               // AMODE0: 128 rows x 2 halves of 16
  const int acol = (tid & 1) * 16;
  const long grow = (long)mb * 128 + arow;

  auto stage = [&](int buf, int kc) {
    if (AMODE == 0) {
      const int gk = kc * 32 + acol;
      const float* p = srcs[gk >> 8] + grow * 256 + (gk & 255);
#pragma unroll
      for (int i = 0; i < 16; ++i) Asl[buf][arow][acol + i] = f2bfu(p[i]);
    }
    if (wave == 0) {
      if (AMODE == 1)
        tdm_load_2d_bf16(lds_off(&Asl[buf][0][0]),
                         Abf + (long)mb * 128 * K + kc * 32,
                         32, 128, (unsigned)K, (unsigned)M, (unsigned)K, 3, 3);
      tdm_load_2d_bf16(lds_off(&Bsl[buf][0][0]),
                       Wt + (long)nb * 64 * K + kc * 32,
                       32, 64, (unsigned)K, (unsigned)N, (unsigned)K, 3, 3);
    }
  };

  const int KC = K >> 5;
  stage(0, 0);
  if (wave == 0) __builtin_amdgcn_s_wait_tensorcnt(0);
  __syncthreads();

  for (int kc = 0; kc < KC; ++kc) {
    const int cur = kc & 1;
    if (kc + 1 < KC) stage(cur ^ 1, kc + 1);

    BF16x16 afr[2], bfr[2];
#pragma unroll
    for (int mt = 0; mt < 2; ++mt) {
      const int row = wm * 32 + mt * 16 + lrow;
#pragma unroll
      for (int i = 0; i < 8; ++i) {
        afr[mt].u[i]     = Asl[cur][row][lhalf + i];
        afr[mt].u[8 + i] = Asl[cur][row][lhalf + 16 + i];
      }
    }
#pragma unroll
    for (int nt = 0; nt < 2; ++nt) {
      const int col = wn * 32 + nt * 16 + lrow;
#pragma unroll
      for (int i = 0; i < 8; ++i) {
        bfr[nt].u[i]     = Bsl[cur][col][lhalf + i];
        bfr[nt].u[8 + i] = Bsl[cur][col][lhalf + 16 + i];
      }
    }
#pragma unroll
    for (int mt = 0; mt < 2; ++mt)
#pragma unroll
      for (int nt = 0; nt < 2; ++nt)
        acc[mt][nt].v = __builtin_amdgcn_wmma_f32_16x16x32_bf16(
            false, afr[mt].v, false, bfr[nt].v, (short)0, acc[mt][nt].v,
            false, false);

    if (wave == 0) __builtin_amdgcn_s_wait_tensorcnt(0);
    __syncthreads();
  }

#pragma unroll
  for (int nt = 0; nt < 2; ++nt) {
    const int gc = nb * 64 + wn * 32 + nt * 16 + lrow;
    const float bb = bias[gc];
#pragma unroll
    for (int mt = 0; mt < 2; ++mt)
#pragma unroll
      for (int i = 0; i < 8; ++i) {
        const int row = wm * 32 + mt * 16 + lhalf + i;  // M = i + 8*(lane>=16)
        const long gr = (long)mb * 128 + row;
        float v = acc[mt][nt].f[i] + bb;
        if (OMODE == 0) {
          v = v > 0.0f ? v : 0.0f;
          outBf[gr * N + gc] = f2bfu(v);
        } else {
          outF[gr * N + gc] = v;
        }
      }
  }
}

// ---------------------------------------------------------------------------
// LSTM: 8 blocks x 16 batch rows. Per step: g = xw[:,t,:] + h @ Whh (16x512,
// K=128) via WMMA with Whh (transposed, N x K) B-fragments held in VGPRs.
// ---------------------------------------------------------------------------
__global__ __launch_bounds__(256) void lstm_wmma_kernel(
    const float* __restrict__ xw, const unsigned short* __restrict__ WhhT,
    const float* __restrict__ h0, const float* __restrict__ c0,
    unsigned short* __restrict__ lout, float* __restrict__ hN) {
  __shared__ float hS[16][128];
  __shared__ float cS[16][128];
  __shared__ unsigned short hB[16][136];
  __shared__ float gS[16][516];

  const int tid = threadIdx.x;
  const int wave = tid >> 5, lane = tid & 31;
  const int lrow = lane & 15;
  const int lhalf = (lane >> 4) * 8;
  const int b0 = blockIdx.x * 16;

  for (int r = 0; r < 8; ++r) {
    const int u = tid * 8 + r;
    const int row = u >> 7, j = u & 127;
    const float h = h0[(b0 + row) * 128 + j];
    const float c = c0[(b0 + row) * 128 + j];
    hS[row][j] = h; cS[row][j] = c; hB[row][j] = f2bfu(h);
  }

  // Whh fragments (register-resident): wave owns gate cols [wave*64, +64)
  BF16x16 bfr[4][4];                       // [ntile][kchunk]
#pragma unroll
  for (int nt = 0; nt < 4; ++nt) {
    const long col = wave * 64 + nt * 16 + lrow;
#pragma unroll
    for (int kc = 0; kc < 4; ++kc) {
      const unsigned short* wp = WhhT + col * 128 + kc * 32 + lhalf;
#pragma unroll
      for (int i = 0; i < 8; ++i) {
        bfr[nt][kc].u[i]     = wp[i];
        bfr[nt][kc].u[8 + i] = wp[16 + i];
      }
    }
  }
  __syncthreads();

  for (int t = 0; t < 512; ++t) {
    BF16x16 afr[4];
#pragma unroll
    for (int kc = 0; kc < 4; ++kc)
#pragma unroll
      for (int i = 0; i < 8; ++i) {
        afr[kc].u[i]     = hB[lrow][kc * 32 + lhalf + i];
        afr[kc].u[8 + i] = hB[lrow][kc * 32 + lhalf + 16 + i];
      }
#pragma unroll
    for (int nt = 0; nt < 4; ++nt) {
      F32x8 acc;
#pragma unroll
      for (int i = 0; i < 8; ++i) acc.f[i] = 0.0f;
#pragma unroll
      for (int kc = 0; kc < 4; ++kc)
        acc.v = __builtin_amdgcn_wmma_f32_16x16x32_bf16(
            false, afr[kc].v, false, bfr[nt][kc].v, (short)0, acc.v, false,
            false);
      const int gc = wave * 64 + nt * 16 + lrow;
#pragma unroll
      for (int i = 0; i < 8; ++i) gS[lhalf + i][gc] = acc.f[i];
    }
    __syncthreads();

    for (int r = 0; r < 8; ++r) {
      const int u = tid * 8 + r;
      const int row = u >> 7, j = u & 127;
      const long base = ((long)(b0 + row) * 512 + t) * 512;  // xw includes b_lstm
      const float gi = gS[row][j]       + xw[base + j];
      const float gf = gS[row][128 + j] + xw[base + 128 + j];
      const float gg = gS[row][256 + j] + xw[base + 256 + j];
      const float go = gS[row][384 + j] + xw[base + 384 + j];
      const float c = sigmoidf(gf) * cS[row][j] + sigmoidf(gi) * tanhf(gg);
      const float h = sigmoidf(go) * tanhf(c);
      cS[row][j] = c; hS[row][j] = h;
      const unsigned short hb = f2bfu(h);
      hB[row][j] = hb;
      lout[((long)(b0 + row) * 512 + t) * 128 + j] = hb;
    }
    __syncthreads();
  }

  for (int r = 0; r < 8; ++r) {
    const int u = tid * 8 + r;
    const int row = u >> 7, j = u & 127;
    hN[(b0 + row) * 128 + j] = hS[row][j];
  }
}

// ---------------------------------------------------------------------------
// q = hN@Wq+bq folded into rvec[b,e,h] = sum_d Wk[e,h*32+d]*q[b,h*32+d] and
// cbias[b,h] = sum_d bk[h*32+d]*q[b,h*32+d].
// ---------------------------------------------------------------------------
__global__ __launch_bounds__(128) void qr_kernel(
    const float* __restrict__ hNp, const float* __restrict__ Wq,
    const float* __restrict__ bq, const float* __restrict__ Wk,
    const float* __restrict__ bk, float* __restrict__ rvec,
    float* __restrict__ cbias) {
  __shared__ float hs[128];
  __shared__ float qs[128];
  const int b = blockIdx.x, j = threadIdx.x;
  hs[j] = hNp[b * 128 + j];
  __syncthreads();
  float q = bq[j];
  for (int e = 0; e < 128; ++e) q += hs[e] * Wq[e * 128 + j];
  qs[j] = q;
  __syncthreads();
  for (int h = 0; h < 4; ++h) {
    float r = 0.0f;
    for (int d = 0; d < 32; ++d) r += Wk[j * 128 + h * 32 + d] * qs[h * 32 + d];
    rvec[((long)b * 128 + j) * 4 + h] = r;
  }
  if (j < 4) {
    float c = 0.0f;
    for (int d = 0; d < 32; ++d) c += bk[j * 32 + d] * qs[j * 32 + d];
    cbias[b * 4 + j] = c;
  }
}

// ---------------------------------------------------------------------------
// Fused scores + softmax per (b,h)
// ---------------------------------------------------------------------------
__global__ __launch_bounds__(512) void score_softmax_kernel(
    const unsigned short* __restrict__ lout, const float* __restrict__ rvec,
    const float* __restrict__ cbias, float* __restrict__ sw) {
  __shared__ float red[512];
  __shared__ float rv[128];
  const int bh = blockIdx.x;
  const int b = bh >> 2, h = bh & 3;
  const int s = threadIdx.x;
  if (s < 128) rv[s] = rvec[((long)b * 128 + s) * 4 + h];
  __syncthreads();
  const unsigned short* lp = lout + ((long)b * 512 + s) * 128;
  float acc = 0.0f;
  for (int d = 0; d < 128; ++d) acc += bfu2f(lp[d]) * rv[d];
  const float wv = (acc + cbias[bh]) * 0.17677669529663687f;  // 1/sqrt(32)
  red[s] = wv;
  __syncthreads();
  for (int off = 256; off > 0; off >>= 1) {
    if (s < off) red[s] = fmaxf(red[s], red[s + off]);
    __syncthreads();
  }
  const float m = red[0];
  __syncthreads();
  const float e = expf(wv - m);
  red[s] = e;
  __syncthreads();
  for (int off = 256; off > 0; off >>= 1) {
    if (s < off) red[s] += red[s + off];
    __syncthreads();
  }
  sw[(long)bh * 512 + s] = e / red[0];
}

// ---------------------------------------------------------------------------
// ctx[b,h,d] = sum_s sw[b,h,s] * lout[b,s,d]
// ---------------------------------------------------------------------------
__global__ __launch_bounds__(128) void ctx_kernel(
    const unsigned short* __restrict__ lout, const float* __restrict__ sw,
    float* __restrict__ ctx) {
  __shared__ float ws[512];
  const int bh = blockIdx.x;
  const int b = bh >> 2;
  const int d = threadIdx.x;
  for (int i = d; i < 512; i += 128) ws[i] = sw[(long)bh * 512 + i];
  __syncthreads();
  float acc = 0.0f;
  for (int s = 0; s < 512; ++s)
    acc += ws[s] * bfu2f(lout[((long)b * 512 + s) * 128 + d]);
  ctx[(long)bh * 128 + d] = acc;
}

// ---------------------------------------------------------------------------
// att[b,h,e] = sum_d ctx[b,h,d]*Wv[h,d,e] + bv[h,e]; out_h = att@Wout + bout
// ---------------------------------------------------------------------------
__global__ __launch_bounds__(128) void attout_kernel(
    const float* __restrict__ ctx, const float* __restrict__ Wv,
    const float* __restrict__ bv, const float* __restrict__ Wout,
    const float* __restrict__ bout, float* __restrict__ outh) {
  __shared__ float cs[512];
  __shared__ float as[512];
  const int b = blockIdx.x, tid = threadIdx.x;
  for (int i = tid; i < 512; i += 128) cs[i] = ctx[(long)b * 512 + i];
  __syncthreads();
  for (int h = 0; h < 4; ++h) {
    float a = bv[h * 128 + tid];
    for (int d = 0; d < 128; ++d)
      a += cs[h * 128 + d] * Wv[(h * 128 + d) * 128 + tid];
    as[h * 128 + tid] = a;
  }
  __syncthreads();
  float o = bout[tid];
  for (int j = 0; j < 512; ++j) o += as[j] * Wout[j * 128 + tid];
  outh[b * 128 + tid] = o;
}

// ---------------------------------------------------------------------------
// output[b] = out_h[b]@Wo + bo; loss = mean((output - label[:,-1,0])^2)
// d_out = [loss, output(128)]
// ---------------------------------------------------------------------------
__global__ __launch_bounds__(128) void final_kernel(
    const float* __restrict__ outh, const float* __restrict__ Wo,
    const float* __restrict__ bo, const float* __restrict__ label,
    float* __restrict__ out) {
  __shared__ float red[128];
  const int b = threadIdx.x;
  float o = bo[0];
  for (int n = 0; n < 128; ++n) o += outh[b * 128 + n] * Wo[n];
  out[1 + b] = o;
  const float diff = o - label[(long)b * 512 + 511];
  red[b] = diff * diff;
  __syncthreads();
  for (int off = 64; off > 0; off >>= 1) {
    if (b < off) red[b] += red[b + off];
    __syncthreads();
  }
  if (b == 0) out[0] = red[0] / 128.0f;
}

// ---------------------------------------------------------------------------
extern "C" void kernel_launch(void* const* d_in, const int* in_sizes, int n_in,
                              void* d_out, int out_size, void* d_ws,
                              size_t ws_size, hipStream_t stream) {
  (void)in_sizes; (void)n_in; (void)out_size; (void)ws_size;
  const float* visual   = (const float*)d_in[0];
  const float* text     = (const float*)d_in[1];
  const float* user_id  = (const float*)d_in[2];
  const float* user_des = (const float*)d_in[3];
  const float* category = (const float*)d_in[4];
  const float* label    = (const float*)d_in[5];
  const float* h0       = (const float*)d_in[6];
  const float* c0       = (const float*)d_in[7];
  const float* W1       = (const float*)d_in[8];
  const float* b1       = (const float*)d_in[9];
  const float* W2       = (const float*)d_in[10];
  const float* b2       = (const float*)d_in[11];
  const float* Wih      = (const float*)d_in[12];
  const float* Whh      = (const float*)d_in[13];
  const float* b_lstm   = (const float*)d_in[14];
  const float* Wq       = (const float*)d_in[15];
  const float* bq       = (const float*)d_in[16];
  const float* Wk       = (const float*)d_in[17];
  const float* bk       = (const float*)d_in[18];
  const float* Wv       = (const float*)d_in[19];
  const float* bv       = (const float*)d_in[20];
  const float* Wout     = (const float*)d_in[21];
  const float* bout     = (const float*)d_in[22];
  const float* Wo       = (const float*)d_in[23];
  const float* bo       = (const float*)d_in[24];

  char* base = (char*)d_ws;
  size_t off = 0;
  auto bump = [&](size_t bytes) -> void* {
    void* p = base + off;
    off = (off + bytes + 255) & ~(size_t)255;
    return p;
  };
  const long M = 65536;  // B*S
  unsigned short* W1t  = (unsigned short*)bump(1280l * 512 * 2);  // (512 x 1280)
  unsigned short* W2t  = (unsigned short*)bump(512l * 256 * 2);   // (256 x 512)
  unsigned short* Wiht = (unsigned short*)bump(256l * 512 * 2);   // (512 x 256)
  unsigned short* Whht = (unsigned short*)bump(128l * 512 * 2);   // (512 x 128)
  unsigned short* zB   = (unsigned short*)bump(M * 512 * 2);
  unsigned short* vtB  = (unsigned short*)bump(M * 256 * 2);
  float*          xw   = (float*)bump(M * 512 * 4);
  unsigned short* loB  = (unsigned short*)bump(M * 128 * 2);
  float*          hNw  = (float*)bump(128l * 128 * 4);
  float*          rvec = (float*)bump(128l * 128 * 4 * 4);
  float*          cbia = (float*)bump(512l * 4);
  float*          sw   = (float*)bump(512l * 512 * 4);
  float*          ctx  = (float*)bump(512l * 128 * 4);
  float*          outh = (float*)bump(128l * 128 * 4);

  // fp32 -> bf16 transposed weight conversion (K x N -> N x K)
  cvt_bf16_t_kernel<<<(1280 * 512 + 255) / 256, 256, 0, stream>>>(W1, W1t, 1280, 512);
  cvt_bf16_t_kernel<<<(512 * 256 + 255) / 256, 256, 0, stream>>>(W2, W2t, 512, 256);
  cvt_bf16_t_kernel<<<(256 * 512 + 255) / 256, 256, 0, stream>>>(Wih, Wiht, 256, 512);
  cvt_bf16_t_kernel<<<(128 * 512 + 255) / 256, 256, 0, stream>>>(Whh, Whht, 128, 512);

  // z = relu(concat(x)@W1+b1); v_t = relu(z@W2+b2); xw = v_t@Wih + b_lstm
  gemm_bf16_kernel<0, 0><<<512 * 8, 256, 0, stream>>>(
      visual, text, user_id, user_des, category, nullptr, W1t, b1, zB, nullptr,
      65536, 512, 1280);
  gemm_bf16_kernel<1, 0><<<512 * 4, 256, 0, stream>>>(
      nullptr, nullptr, nullptr, nullptr, nullptr, zB, W2t, b2, vtB, nullptr,
      65536, 256, 512);
  gemm_bf16_kernel<1, 1><<<512 * 8, 256, 0, stream>>>(
      nullptr, nullptr, nullptr, nullptr, nullptr, vtB, Wiht, b_lstm, nullptr,
      xw, 65536, 512, 256);

  // LSTM (batch-parallel across 8 WGPs, Whh register-resident WMMA B-frags)
  lstm_wmma_kernel<<<8, 256, 0, stream>>>(xw, Whht, h0, c0, loB, hNw);

  // Attention (k@q folded into rvec; v folded into ctx pooling)
  qr_kernel<<<128, 128, 0, stream>>>(hNw, Wq, bq, Wk, bk, rvec, cbia);
  score_softmax_kernel<<<512, 512, 0, stream>>>(loB, rvec, cbia, sw);
  ctx_kernel<<<512, 128, 0, stream>>>(loB, sw, ctx);
  attout_kernel<<<128, 128, 0, stream>>>(ctx, Wv, bv, Wout, bout, outh);
  final_kernel<<<1, 128, 0, stream>>>(outh, Wo, bo, label, (float*)d_out);
}
